// Disc_53515292508892
// MI455X (gfx1250) — compile-verified, matrix operand order checked by reading, not run on hardware
//
#include <hip/hip_runtime.h>
#include <hip/hip_bf16.h>

// Problem dimensions (fixed by the reference).
#define Bdim 64
#define Hdim 1024
#define Ddim 512
#define Tsrc 128
#define Ttgt 128
#define NWG  32          // persistent workgroups per scan (H / 32 column tiles)

typedef __attribute__((ext_vector_type(16))) __bf16 v16bf;
typedef __attribute__((ext_vector_type(8)))  float  v8f;

static __device__ __forceinline__ __bf16 f2bf(float f) {
  union { float f; unsigned u; } v; v.f = f;
  unsigned r = v.u + 0x7FFFu + ((v.u >> 16) & 1u);   // round-to-nearest-even
  unsigned short h = (unsigned short)(r >> 16);
  return __builtin_bit_cast(__bf16, h);
}

// Load one wave-lane's 16 bf16 A/B elements for v_wmma_f32_16x16x32_bf16.
// Per ISA layout: elements 0..7 are K = kA..kA+7 (contiguous), elements
// 8..15 are K = kA+16..kA+23 (contiguous) — two b128 loads.
static __device__ __forceinline__ v16bf load_b16(const __bf16* p) {
  v16bf r;
#pragma unroll
  for (int i = 0; i < 8; ++i) { r[i] = p[i]; r[i + 8] = p[16 + i]; }
  return r;
}

static __device__ __forceinline__ float sigmoidf_(float x) {
  return 1.0f / (1.0f + __expf(-x));
}

__global__ void zero_b32(unsigned* __restrict__ p, int n) {
  int i = blockIdx.x * blockDim.x + threadIdx.x;
  if (i < n) p[i] = 0u;
}

__global__ void cvt_bf16(const float* __restrict__ s, __bf16* __restrict__ d, int n) {
  int i = blockIdx.x * blockDim.x + threadIdx.x;
  if (i < n) d[i] = f2bf(s[i]);
}

// One block per (t,b) row: X[row*D + d] = bf16(emb[ids[row]*D + d])
__global__ void gather_embed(const int* __restrict__ ids, const float* __restrict__ emb,
                             __bf16* __restrict__ X, int rows) {
  int row = blockIdx.x;
  if (row >= rows) return;
  const float* e = emb + (size_t)ids[row] * Ddim;
  __bf16* x = X + (size_t)row * Ddim;
  for (int d = threadIdx.x; d < Ddim; d += blockDim.x) x[d] = f2bf(e[d]);
}

// Persistent GRU scan: one launch runs all T steps. Grid = NWG (=H/32)
// workgroups of 128 threads (4 waves); wave w owns batch rows [16w,16w+16)
// x hidden cols [32*blockIdx.x,+32) (two 16-col WMMA tiles; A reused across
// 6 WMMAs per K-chunk). Steps are separated by a device-scope barrier on
// per-step monotonic counters (zeroed each launch). accR/accZ merge x+h
// preacts; accHN/accXN stay separate because n = tanh(xn + r*hn). Hidden
// state is kept f32 (exact freeze semantics) plus a bf16 shadow written in
// the epilogue so the next step's A operand is two straight b128 loads.
__global__ __launch_bounds__(128) void gru_scan(
    float*  __restrict__ h_a,   float*  __restrict__ h_b,
    __bf16* __restrict__ hbf_a, __bf16* __restrict__ hbf_b,
    const __bf16* __restrict__ X,
    const __bf16* __restrict__ Whh, const __bf16* __restrict__ Wih,
    const float* __restrict__ bih, const float* __restrict__ bhh,
    const int* __restrict__ lengths,
    int* __restrict__ bar, int T)
{
  const int lane = threadIdx.x & 31;
  const int wave = threadIdx.x >> 5;
  const int half = lane >> 4;          // K-half selector for A/B operands
  const int l16  = lane & 15;
  const int c0   = blockIdx.x * 32;    // column tile base in [0,H)
  const int rowA = wave * 16 + l16;    // batch row this lane feeds (A operand)
  const int rw0  = c0 + l16;           // weight row, first N-tile

  // Loop-invariant hoists: gate biases for both N-tiles, per-row lengths.
  float biasR[2], biasZ[2], biasXN[2], biasHN[2];
#pragma unroll
  for (int nt = 0; nt < 2; ++nt) {
    const int col = c0 + nt * 16 + l16;
    biasR[nt]  = bih[col]        + bhh[col];
    biasZ[nt]  = bih[Hdim + col] + bhh[Hdim + col];
    biasXN[nt] = bih[2 * Hdim + col];
    biasHN[nt] = bhh[2 * Hdim + col];
  }
  int lenv[8];
#pragma unroll
  for (int e = 0; e < 8; ++e) lenv[e] = lengths[wave * 16 + e + half * 8];

  // Single base pointer per weight matrix; section / N-tile strides are
  // compile-time constants that fold into the load IOFFSET (max 4.2MB < 8MB).
  const size_t HH = (size_t)Hdim * Hdim;   // Whh gate-section stride (elems)
  const size_t HD = (size_t)Hdim * Ddim;   // Wih gate-section stride (elems)
  const __bf16* wb   = Whh + (size_t)rw0 * Hdim;
  const __bf16* vb   = Wih + (size_t)rw0 * Ddim;
  const __bf16* xrow = X   + (size_t)rowA * Ddim;   // advances B*D per step

  for (int t = 0; t < T; ++t) {
    const float*  h_in    = (t & 1) ? h_b   : h_a;
    float*        h_out   = (t & 1) ? h_a   : h_b;
    const __bf16* hbf_in  = (t & 1) ? hbf_b : hbf_a;
    __bf16*       hbf_out = (t & 1) ? hbf_a : hbf_b;

    v8f accR[2]  = {v8f{}, v8f{}};
    v8f accZ[2]  = {v8f{}, v8f{}};
    v8f accHN[2] = {v8f{}, v8f{}};
    v8f accXN[2] = {v8f{}, v8f{}};

    // ---- hidden part: K = 0..H-1 over Whh (bf16, row-major [3H,H]) ----
    {
      const __bf16* ha = hbf_in + (size_t)rowA * Hdim;
      for (int k = 0; k < Hdim; k += 32) {
        const int kA = k + half * 8;
        __builtin_prefetch(wb + kA + 256, 0, 1);   // global_prefetch_b8
        v16bf a   = load_b16(ha + kA);
        v16bf br0 = load_b16(wb + kA);
        v16bf br1 = load_b16(wb + kA + 16 * Hdim);
        v16bf bz0 = load_b16(wb + kA + HH);
        v16bf bz1 = load_b16(wb + kA + HH + 16 * Hdim);
        v16bf bn0 = load_b16(wb + kA + 2 * HH);
        v16bf bn1 = load_b16(wb + kA + 2 * HH + 16 * Hdim);
        accR[0]  = __builtin_amdgcn_wmma_f32_16x16x32_bf16(false, a, false, br0, (short)0, accR[0],  false, false);
        accR[1]  = __builtin_amdgcn_wmma_f32_16x16x32_bf16(false, a, false, br1, (short)0, accR[1],  false, false);
        accZ[0]  = __builtin_amdgcn_wmma_f32_16x16x32_bf16(false, a, false, bz0, (short)0, accZ[0],  false, false);
        accZ[1]  = __builtin_amdgcn_wmma_f32_16x16x32_bf16(false, a, false, bz1, (short)0, accZ[1],  false, false);
        accHN[0] = __builtin_amdgcn_wmma_f32_16x16x32_bf16(false, a, false, bn0, (short)0, accHN[0], false, false);
        accHN[1] = __builtin_amdgcn_wmma_f32_16x16x32_bf16(false, a, false, bn1, (short)0, accHN[1], false, false);
      }
    }

    // ---- input part: K = 0..D-1 over Wih (bf16, row-major [3H,D]) ----
    for (int k = 0; k < Ddim; k += 32) {
      const int kA = k + half * 8;
      __builtin_prefetch(vb + kA + 256, 0, 1);
      v16bf a   = load_b16(xrow + kA);
      v16bf br0 = load_b16(vb + kA);
      v16bf br1 = load_b16(vb + kA + 16 * Ddim);
      v16bf bz0 = load_b16(vb + kA + HD);
      v16bf bz1 = load_b16(vb + kA + HD + 16 * Ddim);
      v16bf bn0 = load_b16(vb + kA + 2 * HD);
      v16bf bn1 = load_b16(vb + kA + 2 * HD + 16 * Ddim);
      accR[0]  = __builtin_amdgcn_wmma_f32_16x16x32_bf16(false, a, false, br0, (short)0, accR[0],  false, false);
      accR[1]  = __builtin_amdgcn_wmma_f32_16x16x32_bf16(false, a, false, br1, (short)0, accR[1],  false, false);
      accZ[0]  = __builtin_amdgcn_wmma_f32_16x16x32_bf16(false, a, false, bz0, (short)0, accZ[0],  false, false);
      accZ[1]  = __builtin_amdgcn_wmma_f32_16x16x32_bf16(false, a, false, bz1, (short)0, accZ[1],  false, false);
      accXN[0] = __builtin_amdgcn_wmma_f32_16x16x32_bf16(false, a, false, bn0, (short)0, accXN[0], false, false);
      accXN[1] = __builtin_amdgcn_wmma_f32_16x16x32_bf16(false, a, false, bn1, (short)0, accXN[1], false, false);
    }

    // ---- epilogue: gates + masked hidden update (f32 + bf16 shadow) ----
    // C/D layout: VGPR e holds (M = e + 8*half, N = l16).
#pragma unroll
    for (int nt = 0; nt < 2; ++nt) {
      const int col = c0 + nt * 16 + l16;
#pragma unroll
      for (int e = 0; e < 8; ++e) {
        const int   b    = wave * 16 + e + half * 8;
        const float r    = sigmoidf_(accR[nt][e] + biasR[nt]);
        const float z    = sigmoidf_(accZ[nt][e] + biasZ[nt]);
        const float n    = tanhf(accXN[nt][e] + biasXN[nt] + r * (accHN[nt][e] + biasHN[nt]));
        const float hold = h_in[(size_t)b * Hdim + col];
        const float hnew = (1.0f - z) * n + z * hold;
        const float hnx  = (t < lenv[e]) ? hnew : hold;  // freeze past length
        h_out[(size_t)b * Hdim + col]   = hnx;
        hbf_out[(size_t)b * Hdim + col] = f2bf(hnx);
      }
    }

    // ---- device-scope inter-WG barrier (monotonic per-step counter) ----
    __threadfence();        // release: h stores visible at agent scope
    __syncthreads();
    if (threadIdx.x == 0) {
      __hip_atomic_fetch_add(bar + t, 1, __ATOMIC_ACQ_REL, __HIP_MEMORY_SCOPE_AGENT);
      while (__hip_atomic_load(bar + t, __ATOMIC_ACQUIRE, __HIP_MEMORY_SCOPE_AGENT) < NWG) {
        __builtin_amdgcn_s_sleep(2);
      }
    }
    __syncthreads();
    __threadfence();        // acquire: invalidate near caches before reads

    xrow += (size_t)Bdim * Ddim;
  }
}

// logits[b] = tanh(h[b] @ p1W.T + p1b) @ p2W.T + p2b ; one block per batch row.
__global__ __launch_bounds__(64) void proj_head(
    const float* __restrict__ h,
    const float* __restrict__ p1W, const float* __restrict__ p1b,
    const float* __restrict__ p2W, const float* __restrict__ p2b,
    float* __restrict__ out)
{
  __shared__ float s[64];
  const int b = blockIdx.x;
  const int j = threadIdx.x;
  const float* hb = h   + (size_t)b * Hdim;
  const float* w  = p1W + (size_t)j * Hdim;
  float acc = p1b[j];
  for (int k = 0; k < Hdim; ++k) acc += hb[k] * w[k];
  s[j] = tanhf(acc);
  __syncthreads();
  if (j < 2) {
    float o = p2b[j];
    for (int q = 0; q < 64; ++q) o += s[q] * p2W[j * 64 + q];
    out[b * 2 + j] = o;
  }
}

extern "C" void kernel_launch(void* const* d_in, const int* in_sizes, int n_in,
                              void* d_out, int out_size, void* d_ws, size_t ws_size,
                              hipStream_t stream) {
  (void)in_sizes; (void)n_in; (void)out_size; (void)ws_size;

  const int*   tgt     = (const int*)d_in[0];
  const int*   tgt_len = (const int*)d_in[1];
  const int*   src     = (const int*)d_in[2];
  const int*   src_len = (const int*)d_in[3];
  // d_in[4], d_in[5]: ref_ids / ref_lengths — ref branch is dead code in the reference
  const float* emb     = (const float*)d_in[6];
  const float* sWih    = (const float*)d_in[7];
  const float* sWhh    = (const float*)d_in[8];
  const float* sbih    = (const float*)d_in[9];
  const float* sbhh    = (const float*)d_in[10];
  const float* tWih    = (const float*)d_in[11];
  const float* tWhh    = (const float*)d_in[12];
  const float* tbih    = (const float*)d_in[13];
  const float* tbhh    = (const float*)d_in[14];
  // d_in[15..18]: ref GRU params — unused
  const float* p1_W    = (const float*)d_in[19];
  const float* p1_b    = (const float*)d_in[20];
  const float* p2_W    = (const float*)d_in[21];
  const float* p2_b    = (const float*)d_in[22];
  float* out = (float*)d_out;

  // ---- workspace carve-out (all initialized every call; ~37 MB) ----
  char* ws = (char*)d_ws;
  size_t off = 0;
  auto alloc = [&](size_t bytes) -> char* {
    char* p = ws + off;
    off = (off + bytes + 255) & ~(size_t)255;
    return p;
  };
  float*  h_a    = (float*) alloc((size_t)Bdim * Hdim * sizeof(float));
  float*  h_b    = (float*) alloc((size_t)Bdim * Hdim * sizeof(float));
  __bf16* hb_a   = (__bf16*)alloc((size_t)Bdim * Hdim * sizeof(__bf16));
  __bf16* hb_b   = (__bf16*)alloc((size_t)Bdim * Hdim * sizeof(__bf16));
  int*    bar    = (int*)   alloc((size_t)(Tsrc + Ttgt) * sizeof(int));
  __bf16* Xsrc   = (__bf16*)alloc((size_t)Tsrc * Bdim * Ddim * sizeof(__bf16));
  __bf16* Xtgt   = (__bf16*)alloc((size_t)Ttgt * Bdim * Ddim * sizeof(__bf16));
  __bf16* sWhh_b = (__bf16*)alloc((size_t)3 * Hdim * Hdim * sizeof(__bf16));
  __bf16* sWih_b = (__bf16*)alloc((size_t)3 * Hdim * Ddim * sizeof(__bf16));
  __bf16* tWhh_b = (__bf16*)alloc((size_t)3 * Hdim * Hdim * sizeof(__bf16));
  __bf16* tWih_b = (__bf16*)alloc((size_t)3 * Hdim * Ddim * sizeof(__bf16));

  // ---- prep: zero h0 (f32+bf16) and barrier counters, gather, convert ----
  // h_a (f32) + h_b-sized hole skipped; zero exactly what step 0 reads.
  zero_b32<<<(Bdim * Hdim + 255) / 256, 256, 0, stream>>>((unsigned*)h_a, Bdim * Hdim);
  zero_b32<<<(Bdim * Hdim / 2 + 255) / 256, 256, 0, stream>>>((unsigned*)hb_a, Bdim * Hdim / 2);
  zero_b32<<<1, 256, 0, stream>>>((unsigned*)bar, Tsrc + Ttgt);
  gather_embed<<<Tsrc * Bdim, 128, 0, stream>>>(src, emb, Xsrc, Tsrc * Bdim);
  gather_embed<<<Ttgt * Bdim, 128, 0, stream>>>(tgt, emb, Xtgt, Ttgt * Bdim);
  cvt_bf16<<<(3 * Hdim * Hdim + 255) / 256, 256, 0, stream>>>(sWhh, sWhh_b, 3 * Hdim * Hdim);
  cvt_bf16<<<(3 * Hdim * Ddim + 255) / 256, 256, 0, stream>>>(sWih, sWih_b, 3 * Hdim * Ddim);
  cvt_bf16<<<(3 * Hdim * Hdim + 255) / 256, 256, 0, stream>>>(tWhh, tWhh_b, 3 * Hdim * Hdim);
  cvt_bf16<<<(3 * Hdim * Ddim + 255) / 256, 256, 0, stream>>>(tWih, tWih_b, 3 * Hdim * Ddim);

  // ---- src encoder: one persistent kernel scans all 128 steps ----
  gru_scan<<<NWG, 128, 0, stream>>>(h_a, h_b, hb_a, hb_b, Xsrc,
                                    sWhh_b, sWih_b, sbih, sbhh, src_len,
                                    bar, Tsrc);
  // T=128 (even) -> final src hidden lands back in h_a / hb_a.

  // ---- tgt decoder: h0 = src final hidden, same buffers ----
  gru_scan<<<NWG, 128, 0, stream>>>(h_a, h_b, hb_a, hb_b, Xtgt,
                                    tWhh_b, tWih_b, tbih, tbhh, tgt_len,
                                    bar + Tsrc, Ttgt);

  // ---- projection head ----
  proj_head<<<Bdim, 64, 0, stream>>>(h_a, p1_W, p1_b, p2_W, p2_b, out);
}
